// UpsamplingModule_5549097746946
// MI455X (gfx1250) — matrix operand order repaired and checked by reading, not compiled, not run on hardware
//
#include <hip/hip_runtime.h>
#include <stdint.h>

// ---------------------------------------------------------------------------
// Types for CDNA5 WMMA / TDM (gfx1250, wave32)
// ---------------------------------------------------------------------------
typedef __attribute__((ext_vector_type(16))) __bf16 v16bf;
typedef __attribute__((ext_vector_type(8)))  __bf16 v8bf;
typedef __attribute__((ext_vector_type(8)))  float  v8f;
typedef __attribute__((ext_vector_type(4)))  unsigned int u32x4;
typedef __attribute__((ext_vector_type(4)))  int    i32x4;
typedef __attribute__((ext_vector_type(8)))  int    i32x8;

#if defined(__has_builtin)
#if __has_builtin(__builtin_amdgcn_tensor_load_to_lds) && __has_builtin(__builtin_amdgcn_s_wait_tensorcnt)
#define USE_TDM 1
#endif
#endif

static __device__ __forceinline__ v8f wmma_bf16(v16bf a, v16bf b, v8f c) {
  // D = A(16x32 bf16) * B(32x16 bf16) + C(16x16 f32)
  return __builtin_amdgcn_wmma_f32_16x16x32_bf16(
      /*neg_a=*/false, a, /*neg_b=*/false, b,
      /*c_mod=*/(short)0, c, /*reuse_a=*/false, /*reuse_b=*/false);
}

// A-matrix (16x32, bf16) lane layout (contiguous k, matches B):
//   lanes 0-15 : row M = lane,    elements = K 0..15
//   lanes 16-31: row M = lane-16, elements = K 16..31
static __device__ __forceinline__ v16bf loadA16x32(const __bf16* rowptr, int lane) {
  const int base = (lane < 16) ? 0 : 16;
  union { v16bf v; v8bf h[2]; } u;
  u.h[0] = *(const v8bf*)(rowptr + base);
  u.h[1] = *(const v8bf*)(rowptr + base + 8);
  return u.v;
}

// B-matrix (32x16, bf16): lane = column (mod 16); lanes 0-15 hold K 0..15,
// lanes 16-31 hold K 16..31 -> one contiguous 32-byte load per lane.
static __device__ __forceinline__ v16bf loadB32x16(const __bf16* p) {
  return *(const v16bf*)p;
}

static __device__ __forceinline__ float lrelu(float x) {
  return x > 0.0f ? x : 0.2f * x;
}

// ---------------------------------------------------------------------------
// TDM: 1-D async copy global -> LDS via Tensor Data Mover (ISA ch.8, D# §8.3-8.4)
//   data_size = 8 bytes; nbytes must be a multiple of 8.
// ---------------------------------------------------------------------------
#ifdef USE_TDM
static __device__ __forceinline__ void tdm_copy_to_lds(const void* gptr, void* lptr,
                                                       unsigned int nbytes) {
  unsigned long long ga = (unsigned long long)(uintptr_t)gptr;
  unsigned int la = (unsigned int)(uintptr_t)lptr;  // low 32 bits = LDS byte offset
  unsigned int n8 = nbytes >> 3;                    // elements of 8 bytes

  u32x4 g0;
  g0[0] = 1u;                                          // count=1 (valid D#)
  g0[1] = la;                                          // lds_addr [63:32]
  g0[2] = (unsigned int)(ga & 0xFFFFFFFFu);            // global_addr[31:0]
  g0[3] = ((unsigned int)(ga >> 32) & 0x01FFFFFFu)     // global_addr[56:32]
          | 0x80000000u;                               // type=2 ("image") [127:126]

  i32x8 g1;
  g1[0] = (int)(3u << 16);                             // data_size=3 (8B); wg_mask=0
  g1[1] = (int)((n8 & 0xFFFFu) << 16);                 // tensor_dim0[15:0] @ bit48
  g1[2] = (int)(((n8 >> 16) & 0xFFFFu)                 // tensor_dim0[31:16]
          | (1u << 16));                               // tensor_dim1 = 1
  g1[3] = (int)((n8 & 0xFFFFu) << 16);                 // tile_dim0 @ [127:112]
  g1[4] = 0;                                           // tile_dim1=0, tile_dim2=0
  g1[5] = (int)n8;                                     // tensor_dim0_stride[31:0]
  g1[6] = 0;
  g1[7] = 0;

  i32x4 z4 = {0, 0, 0, 0};
#if __clang_major__ >= 23
  i32x8 z8 = {0, 0, 0, 0, 0, 0, 0, 0};
  __builtin_amdgcn_tensor_load_to_lds(g0, g1, z4, z4, z8, 0);
#else
  __builtin_amdgcn_tensor_load_to_lds(g0, g1, z4, z4, 0);
#endif
}
#endif

// ---------------------------------------------------------------------------
// Constants
// ---------------------------------------------------------------------------
#define NPTS 8192
#define NB   2

// ---------------------------------------------------------------------------
// K0: pack edge-MLP weight [128, Cin] (row-major f32) into WMMA-B layout
// ---------------------------------------------------------------------------
__global__ void k_pack(const float* __restrict__ src, __bf16* __restrict__ dst, int Cin) {
  int i = blockIdx.x * blockDim.x + threadIdx.x;
  int total = Cin * 128;
  if (i >= total) return;
  int kk = i & 31;
  int n  = (i >> 5) & 15;
  int o  = (i >> 9) & 7;
  int kt = i >> 12;
  dst[i] = (__bf16)src[(size_t)(o * 16 + n) * Cin + kt * 32 + kk];
}

// ---------------------------------------------------------------------------
// K1: conv1x1 128->32 + LeakyReLU. One wave per point; lane = out channel.
// ---------------------------------------------------------------------------
__global__ void k_conv128to32(const float* __restrict__ in,
                              const float* __restrict__ W,
                              const float* __restrict__ bias,
                              __bf16* __restrict__ xh,
                              float* __restrict__ sq,
                              long sb, long sn, long sc, int totalPoints) {
  int lane = threadIdx.x & 31;
  int warp = threadIdx.x >> 5;
  int g = blockIdx.x * 8 + warp;
  if (g >= totalPoints) return;
  int b = g / NPTS;
  int n = g % NPTS;
  const float* base = in + (size_t)b * sb + (size_t)n * sn;
  const float* wr   = W + (size_t)lane * 128;
  float acc = bias[lane];
  #pragma unroll 8
  for (int c = 0; c < 128; ++c)
    acc = fmaf(base[(size_t)c * sc], wr[c], acc);
  acc = lrelu(acc);
  __bf16 h = (__bf16)acc;
  float vr = (float)h;
  xh[(size_t)g * 32 + lane] = h;
  float s = vr * vr;
  #pragma unroll
  for (int off = 16; off; off >>= 1) s += __shfl_xor(s, off, 32);
  if (lane == 0) sq[g] = s;
}

// ---------------------------------------------------------------------------
// K2: fused Gram matrix (bf16 WMMA) + streaming top-K kNN.
//     Software-pipelined B loads; all 32 lanes scan (halves of each row).
//     Two-phase scan: vectorized distance + tile-min first; the (rare)
//     insertion chain exists once, inside a non-unrolled loop.
// ---------------------------------------------------------------------------
template <int K>
__global__ __launch_bounds__(32) void k_knn(const __bf16* __restrict__ xh,
                                            const float*  __restrict__ sq,
                                            int* __restrict__ idxOut) {
  const int lane    = threadIdx.x;
  const int b       = blockIdx.y;
  const int rowBase = blockIdx.x * 16;
  const __bf16* xb  = xh + (size_t)b * NPTS * 32;
  const float*  sqb = sq + (size_t)b * NPTS;

  __shared__ __align__(32) float tile[16][17];   // padded
  __shared__ __align__(32) float sqc[16];
  __shared__ float mD[32][K];
  __shared__ int   mI[32][K];

  const v16bf A = loadA16x32(xb + (size_t)(rowBase + (lane & 15)) * 32, lane);

  float bd[K];
  int   bi[K];
  #pragma unroll
  for (int i = 0; i < K; ++i) { bd[i] = 3.4e38f; bi[i] = 0x7fffffff; }

  const int col   = lane & 15;
  const int rb    = (lane < 16) ? 0 : 8;
  const int r     = lane & 15;            // scan row
  const int jbase = (lane < 16) ? 0 : 8;  // scan column half
  const int bsel  = (lane < 16) ? 0 : 16;

  v16bf Bcur = loadB32x16(xb + (size_t)col * 32 + bsel);

  for (int ct = 0; ct < NPTS / 16; ++ct) {
    const int colBase = ct * 16;
    if (lane < 16) sqc[lane] = sqb[colBase + lane];

    v16bf Bnext = Bcur;
    if (ct + 1 < NPTS / 16) {
      Bnext = loadB32x16(xb + (size_t)(colBase + 16 + col) * 32 + bsel);
      __builtin_prefetch(xb + (size_t)(colBase + 32 + col) * 32, 0, 3);
    }

    v8f c = {};
    c = wmma_bf16(A, Bcur, c);
    #pragma unroll
    for (int v = 0; v < 8; ++v) tile[rb + v][col] = c[v];
    __syncthreads();

    // ---- phase 1: vectorized distances + tile minimum (common path) ----
    float d0, d1;
    {
      float dv[8];
      #pragma unroll
      for (int j = 0; j < 8; ++j)
        dv[j] = sqc[jbase + j] - 2.0f * tile[r][jbase + j];
      float m01 = fminf(dv[0], dv[1]), m23 = fminf(dv[2], dv[3]);
      float m45 = fminf(dv[4], dv[5]), m67 = fminf(dv[6], dv[7]);
      d0 = fminf(fminf(m01, m23), fminf(m45, m67));
      d1 = 0.0f; (void)d1;
    }

    // ---- phase 2: rare insertion (single non-unrolled chain) ----
    if (d0 < bd[K - 1]) {
      #pragma unroll 1
      for (int j = 0; j < 8; ++j) {
        const int jj = jbase + j;
        float d = sqc[jj] - 2.0f * tile[r][jj];
        if (d < bd[K - 1]) {
          float td = d; int ti = colBase + jj;
          #pragma unroll
          for (int p = 0; p < K; ++p) {
            if (td < bd[p]) {
              float od = bd[p]; int oi = bi[p];
              bd[p] = td; bi[p] = ti; td = od; ti = oi;
            }
          }
        }
      }
    }
    __syncthreads();
    Bcur = Bnext;
  }

  // merge the two sorted half-lists of each row
  #pragma unroll
  for (int i = 0; i < K; ++i) { mD[lane][i] = bd[i]; mI[lane][i] = bi[i]; }
  __syncthreads();
  if (lane < 16) {
    int* op = idxOut + ((size_t)b * NPTS + rowBase + lane) * K;
    int a = 0, c2 = 0;
    #pragma unroll
    for (int i = 0; i < K; ++i) {
      bool takeA = (c2 >= K) || (a < K && mD[lane][a] <= mD[lane + 16][c2]);
      if (takeA) { op[i] = mI[lane][a]; ++a; }
      else       { op[i] = mI[lane + 16][c2]; ++c2; }
    }
  }
}

// ---------------------------------------------------------------------------
// K3: fused EdgeConv MLP (two bf16 WMMA GEMMs) + max over k.
//     Weights DMA'd into LDS once per block by the Tensor Data Mover.
// ---------------------------------------------------------------------------
template <int K>
__global__ __launch_bounds__(32) void k_edge(const __bf16* __restrict__ xh,
                                             const int*    __restrict__ idx,
                                             const __bf16* __restrict__ WaB,
                                             const float*  __restrict__ ba,
                                             const __bf16* __restrict__ WbB,
                                             const float*  __restrict__ bb,
                                             float* __restrict__ yOut) {
  constexpr int P = 4;
  constexpr int R = P * K;      // 48 (K=12) or 16 (K=4)
  constexpr int T = R / 16;

  const int lane = threadIdx.x;
  const int b    = blockIdx.y;
  const int p0   = blockIdx.x * P;
  const __bf16* xb = xh + (size_t)b * NPTS * 32;

  __shared__ __align__(64) __bf16 sWa[64 * 128];    // 16 KB, WMMA-B layout
  __shared__ __align__(64) __bf16 sWb[128 * 128];   // 32 KB, WMMA-B layout
  __shared__ __align__(32) __bf16 ebuf[16][64];
  __shared__ __align__(32) __bf16 h1buf[16][128];
  __shared__ __align__(32) float  hout[R][128];

#ifdef USE_TDM
  tdm_copy_to_lds(WaB, sWa, 64 * 128 * sizeof(__bf16));
  tdm_copy_to_lds(WbB, sWb, 128 * 128 * sizeof(__bf16));
#else
  {
    const unsigned long long* gA = (const unsigned long long*)WaB;
    unsigned long long*       lA = (unsigned long long*)sWa;
    for (int i = lane; i < 64 * 128 / 4; i += 32) lA[i] = gA[i];
    const unsigned long long* gB = (const unsigned long long*)WbB;
    unsigned long long*       lB = (unsigned long long*)sWb;
    for (int i = lane; i < 128 * 128 / 4; i += 32) lB[i] = gB[i];
  }
  __syncthreads();
#endif

  const int col  = lane & 15;
  const int rb   = (lane < 16) ? 0 : 8;
  const int wsel = (lane < 16) ? 0 : 16;

  for (int t = 0; t < T; ++t) {
    // ---- build edge-feature tile: row r = (point, neighbor) pair ----
    {
      int r  = lane & 15;
      int h  = lane >> 4;                          // 0: x_j - x_i   1: x_i
      int g  = t * 16 + r;
      int p  = p0 + g / K;
      int kk = g % K;
      int nbr = idx[((size_t)b * NPTS + p) * K + kk];
      const __bf16* xp = xb + (size_t)p   * 32;
      const __bf16* xn = xb + (size_t)nbr * 32;
      union { v8bf v[4]; __bf16 e[32]; } u;
      if (h == 0) {
        #pragma unroll
        for (int c = 0; c < 32; ++c)
          u.e[c] = (__bf16)((float)xn[c] - (float)xp[c]);
      } else {
        #pragma unroll
        for (int c = 0; c < 32; ++c) u.e[c] = xp[c];
      }
      v8bf* dst = (v8bf*)&ebuf[r][h * 32];
      #pragma unroll
      for (int q = 0; q < 4; ++q) dst[q] = u.v[q];
    }
    __syncthreads();

#ifdef USE_TDM
    __builtin_amdgcn_s_wait_tensorcnt((short)0);   // weights resident (no-op after t=0)
#endif

    // ---- GEMM1: e[16x64] * Wa^T[64x128] + ba, LeakyReLU -> h1buf(bf16) ----
    {
      v16bf A0 = loadA16x32(&ebuf[lane & 15][0],  lane);
      v16bf A1 = loadA16x32(&ebuf[lane & 15][32], lane);
      #pragma unroll
      for (int o = 0; o < 8; ++o) {
        v16bf B0 = loadB32x16(sWa + ((size_t)((0 * 8 + o) * 16 + col) * 32) + wsel);
        v16bf B1 = loadB32x16(sWa + ((size_t)((1 * 8 + o) * 16 + col) * 32) + wsel);
        v8f c = {};
        c = wmma_bf16(A0, B0, c);
        c = wmma_bf16(A1, B1, c);
        float bv = ba[o * 16 + col];
        #pragma unroll
        for (int v = 0; v < 8; ++v)
          h1buf[rb + v][o * 16 + col] = (__bf16)lrelu(c[v] + bv);
      }
    }
    __syncthreads();

    // ---- GEMM2: h1[16x128] * Wb^T[128x128] + bb, LeakyReLU -> hout(f32) ----
    {
      v16bf Ah[4];
      #pragma unroll
      for (int kt = 0; kt < 4; ++kt)
        Ah[kt] = loadA16x32(&h1buf[lane & 15][32 * kt], lane);
      #pragma unroll
      for (int o = 0; o < 8; ++o) {
        v8f c = {};
        #pragma unroll
        for (int kt = 0; kt < 4; ++kt) {
          v16bf Bw = loadB32x16(sWb + ((size_t)((kt * 8 + o) * 16 + col) * 32) + wsel);
          c = wmma_bf16(Ah[kt], Bw, c);
        }
        float bv = bb[o * 16 + col];
        #pragma unroll
        for (int v = 0; v < 8; ++v)
          hout[t * 16 + rb + v][o * 16 + col] = lrelu(c[v] + bv);
      }
    }
    __syncthreads();
  }

  // ---- max over K neighbors per point, store [B,N,128] ----
  for (int i = lane; i < P * 128; i += 32) {
    int p  = i >> 7;
    int ch = i & 127;
    float m = -3.4e38f;
    #pragma unroll
    for (int kk = 0; kk < K; ++kk) m = fmaxf(m, hout[p * K + kk][ch]);
    yOut[((size_t)b * NPTS + p0 + p) * 128 + ch] = m;
  }
}

// ---------------------------------------------------------------------------
// K4: decoder 128 -> 6 -> 12 -> 12 (tiny, one thread per point)
// ---------------------------------------------------------------------------
__global__ void k_decoder(const float* __restrict__ y,
                          const float* __restrict__ Wd0, const float* __restrict__ bd0,
                          const float* __restrict__ Wd1, const float* __restrict__ bd1,
                          const float* __restrict__ Wd2, const float* __restrict__ bd2,
                          float* __restrict__ out, int total) {
  int g = blockIdx.x * blockDim.x + threadIdx.x;
  if (g >= total) return;
  const float* yp = y + (size_t)g * 128;
  float h0[6];
  #pragma unroll
  for (int o = 0; o < 6; ++o) {
    float s = bd0[o];
    #pragma unroll 8
    for (int c = 0; c < 128; ++c) s = fmaf(yp[c], Wd0[o * 128 + c], s);
    h0[o] = lrelu(s);
  }
  float h1[12];
  #pragma unroll
  for (int o = 0; o < 12; ++o) {
    float s = bd1[o];
    #pragma unroll
    for (int c = 0; c < 6; ++c) s = fmaf(h0[c], Wd1[o * 6 + c], s);
    h1[o] = lrelu(s);
  }
  float* op = out + (size_t)g * 12;
  #pragma unroll
  for (int o = 0; o < 12; ++o) {
    float s = bd2[o];
    #pragma unroll
    for (int c = 0; c < 12; ++c) s = fmaf(h1[c], Wd2[o * 12 + c], s);
    op[o] = s;
  }
}

// ---------------------------------------------------------------------------
// Launcher
// ---------------------------------------------------------------------------
extern "C" void kernel_launch(void* const* d_in, const int* in_sizes, int n_in,
                              void* d_out, int out_size, void* d_ws, size_t ws_size,
                              hipStream_t stream) {
  (void)in_sizes; (void)n_in; (void)out_size; (void)ws_size;

  const float* feature = (const float*)d_in[0];
  const float* Wc0  = (const float*)d_in[1];
  const float* bc0  = (const float*)d_in[2];
  const float* We0a = (const float*)d_in[3];
  const float* be0a = (const float*)d_in[4];
  const float* We0b = (const float*)d_in[5];
  const float* be0b = (const float*)d_in[6];
  const float* Wc1  = (const float*)d_in[7];
  const float* bc1  = (const float*)d_in[8];
  const float* We1a = (const float*)d_in[9];
  const float* be1a = (const float*)d_in[10];
  const float* We1b = (const float*)d_in[11];
  const float* be1b = (const float*)d_in[12];
  const float* Wd0  = (const float*)d_in[13];
  const float* bd0  = (const float*)d_in[14];
  const float* Wd1  = (const float*)d_in[15];
  const float* bd1  = (const float*)d_in[16];
  const float* Wd2  = (const float*)d_in[17];
  const float* bd2  = (const float*)d_in[18];

  char* w = (char*)d_ws;
  auto carve = [&](size_t bytes) -> void* {
    void* p = (void*)w;
    w += (bytes + 255) & ~(size_t)255;
    return p;
  };
  const size_t TP = (size_t)NB * NPTS;   // 16384 points
  __bf16* x0h  = (__bf16*)carve(TP * 32 * sizeof(__bf16));
  float*  sq0  = (float*) carve(TP * sizeof(float));
  int*    idx0 = (int*)   carve(TP * 12 * sizeof(int));
  float*  y0   = (float*) carve(TP * 128 * sizeof(float));
  __bf16* x1h  = (__bf16*)carve(TP * 32 * sizeof(__bf16));
  float*  sq1  = (float*) carve(TP * sizeof(float));
  int*    idx1 = (int*)   carve(TP * 4 * sizeof(int));
  float*  y1   = (float*) carve(TP * 128 * sizeof(float));
  __bf16* WaB0 = (__bf16*)carve((size_t)64  * 128 * sizeof(__bf16));
  __bf16* WbB0 = (__bf16*)carve((size_t)128 * 128 * sizeof(__bf16));
  __bf16* WaB1 = (__bf16*)carve((size_t)64  * 128 * sizeof(__bf16));
  __bf16* WbB1 = (__bf16*)carve((size_t)128 * 128 * sizeof(__bf16));

  // 0) weight pre-swizzle into WMMA B-layout (bf16)
  k_pack<<<(64  * 128 + 255) / 256, 256, 0, stream>>>(We0a, WaB0, 64);
  k_pack<<<(128 * 128 + 255) / 256, 256, 0, stream>>>(We0b, WbB0, 128);
  k_pack<<<(64  * 128 + 255) / 256, 256, 0, stream>>>(We1a, WaB1, 64);
  k_pack<<<(128 * 128 + 255) / 256, 256, 0, stream>>>(We1b, WbB1, 128);

  // 1) conv0 (input NCHW: [B,128,N]) -> x0h bf16 + sq0
  k_conv128to32<<<(int)(TP / 8), 256, 0, stream>>>(
      feature, Wc0, bc0, x0h, sq0,
      (long)128 * NPTS, 1L, (long)NPTS, (int)TP);

  // 2) kNN k=12 via WMMA Gram + streaming top-k
  k_knn<12><<<dim3(NPTS / 16, NB), 32, 0, stream>>>(x0h, sq0, idx0);

  // 3) EdgeConv 0 (k=12): 32 -> 128
  k_edge<12><<<dim3(NPTS / 4, NB), 32, 0, stream>>>(x0h, idx0, WaB0, be0a, WbB0, be0b, y0);

  // 4) conv1 (input [B,N,128]) -> x1h bf16 + sq1
  k_conv128to32<<<(int)(TP / 8), 256, 0, stream>>>(
      y0, Wc1, bc1, x1h, sq1,
      (long)NPTS * 128, 128L, 1L, (int)TP);

  // 5) kNN k=4
  k_knn<4><<<dim3(NPTS / 16, NB), 32, 0, stream>>>(x1h, sq1, idx1);

  // 6) EdgeConv 1 (k=4): 32 -> 128
  k_edge<4><<<dim3(NPTS / 4, NB), 32, 0, stream>>>(x1h, idx1, WaB1, be1a, WbB1, be1b, y1);

  // 7) decoder -> d_out [B, N, 12]
  k_decoder<<<(int)(TP / 256), 256, 0, stream>>>(
      y1, Wd0, bd0, Wd1, bd1, Wd2, bd2, (float*)d_out, (int)TP);
}
